// InteractionBlock_39393440039006
// MI455X (gfx1250) — matrix operand order repaired and checked by reading, not compiled, not run on hardware
//
#include <hip/hip_runtime.h>
#include <hip/hip_bf16.h>

typedef __attribute__((ext_vector_type(16))) _Float16 v16h;
typedef __attribute__((ext_vector_type(8)))  float    v8f;

#define LOG2F_ 0.6931471805599453f

__device__ __forceinline__ float ssp_f(float x) {
  float ax = __builtin_fabsf(x);
  return fmaxf(x, 0.0f) + log1pf(__expf(-ax)) - LOG2F_;
}

union FragU { v16h v; uint32_t u[8]; };

__device__ __forceinline__ uint32_t pack2(float a, float b) {
  union { _Float16 h[2]; uint32_t u; } p;
  p.h[0] = (_Float16)a; p.h[1] = (_Float16)b;
  return p.u;
}

// A fragment (16x32 f16): lane l holds row m=l&15; K pairs contiguous:
//   pair j at k = 16*(l>=16) + 2*j
__device__ __forceinline__ v16h frag_a_lds(const _Float16* tile, int strideH, int kOff) {
  const int l = threadIdx.x & 31;
  const int m = l & 15;
  const int hi8 = (l & 16) ? 16 : 0;
  const uint32_t* row = (const uint32_t*)(tile + m * strideH + kOff);
  FragU r;
#pragma unroll
  for (int j = 0; j < 8; ++j) {
    int k = hi8 + 2 * j;
    r.u[j] = row[k >> 1];
  }
  return r.v;
}

// B fragment (32x16 f16): lane l holds column n = nBase + (l&15);
// lanes 0-15 K=0..15, lanes 16-31 K=16..31. Weights staged in LDS as [n][k]
// so the 16 halves are contiguous -> 8 consecutive u32 LDS reads.
__device__ __forceinline__ v16h frag_b_lds(const _Float16* w, int strideH, int nBase, int kOff) {
  const int l = threadIdx.x & 31;
  const int n = nBase + (l & 15);
  const int hi16 = (l & 16) ? 16 : 0;
  const uint32_t* p = (const uint32_t*)(w + n * strideH + kOff + hi16);
  FragU r;
#pragma unroll
  for (int j = 0; j < 8; ++j) r.u[j] = p[j];
  return r.v;
}

__device__ __forceinline__ v8f wmma_f16(v16h a, v16h b, v8f c) {
  return __builtin_amdgcn_wmma_f32_16x16x32_f16(false, a, false, b, (short)0, c, false, false);
}

// ---------------------------------------------------------------- kernel 1
__global__ void zero_kernel(float* p, long n) {
  long i = blockIdx.x * (long)blockDim.x + threadIdx.x;
  long stride = (long)gridDim.x * blockDim.x;
  for (; i < n; i += stride) p[i] = 0.0f;
}

// ---------------------------------------------------------------- kernel 2
// h = x @ lin1_w^T, stored FEATURE-MAJOR as f16: hbuf[n*N + node].
// Feature-major makes each lane's 8 D-fragment rows contiguous -> b128 stores.
__global__ void node_proj_kernel(const float* __restrict__ x,
                                 const float* __restrict__ lin1_w,
                                 _Float16* __restrict__ hbuf, int N) {
  extern __shared__ char smem[];
  _Float16* sW = (_Float16*)smem;                   // [128][128] = lin1_w as-is
  _Float16* sX = (_Float16*)(smem + 128 * 128 * 2); // 8 waves * [16][128]
  const int tid = threadIdx.x;
  for (int idx = tid; idx < 128 * 128; idx += blockDim.x)
    sW[idx] = (_Float16)lin1_w[idx];
  __syncthreads();
  const int w = tid >> 5, l = tid & 31;
  const int tiles = (N + 15) >> 4;
  int t = blockIdx.x * 8 + w;
  if (t >= tiles) return;
  _Float16* xs = sX + w * (16 * 128);
  const int r = l & 15, sel = (l >> 4) & 1;
  int node = t * 16 + r; if (node >= N) node = N - 1;
  const float4* src = (const float4*)(x + (size_t)node * 128 + sel * 64);
#pragma unroll
  for (int i = 0; i < 16; ++i) {
    float4 v = src[i];
    union { uint32_t u[2]; uint2 q; } p;
    p.u[0] = pack2(v.x, v.y);
    p.u[1] = pack2(v.z, v.w);
    *(uint2*)(xs + r * 128 + sel * 64 + i * 4) = p.q;
  }
  asm volatile("" ::: "memory");
  v16h aK[4];
#pragma unroll
  for (int ks = 0; ks < 4; ++ks) aK[ks] = frag_a_lds(xs, 128, ks * 32);
  const int mOff = (l & 16) ? 8 : 0;
  const int nl = l & 15;
  const bool full = (t * 16 + 16 <= N) && ((N & 7) == 0); // uniform fast path
#pragma unroll
  for (int nt = 0; nt < 8; ++nt) {
    v8f acc;
#pragma unroll
    for (int j = 0; j < 8; ++j) acc[j] = 0.0f;
#pragma unroll
    for (int ks = 0; ks < 4; ++ks)
      acc = wmma_f16(aK[ks], frag_b_lds(sW, 128, nt * 16, ks * 32), acc);
    int n = nt * 16 + nl;
    union { _Float16 h[8]; uint4 q; } o;
#pragma unroll
    for (int j = 0; j < 8; ++j) o.h[j] = (_Float16)acc[j];
    if (full) {
      *(uint4*)(hbuf + (size_t)n * N + t * 16 + mOff) = o.q; // 8 contiguous halves
    } else {
#pragma unroll
      for (int j = 0; j < 8; ++j) {
        int nd = t * 16 + mOff + j;
        if (nd < N) hbuf[(size_t)n * N + nd] = o.h[j];
      }
    }
  }
}

// ---------------------------------------------------------------- kernel 3
// Fused: W = (ssp(attr@W1^T+b1)@W2^T+b2)*C ; msg = h[src]*W ; atomic scatter to agg[dst]
__global__ void edge_kernel(const float* __restrict__ ea, const int* __restrict__ ei,
                            const float* __restrict__ ew,
                            const float* __restrict__ w1, const float* __restrict__ b1,
                            const float* __restrict__ w2, const float* __restrict__ b2,
                            const _Float16* __restrict__ hbuf, float* __restrict__ agg,
                            int E, int N) {
  extern __shared__ char smem[];
  _Float16* sW1 = (_Float16*)smem;                                  // [128][64] (50 padded)
  _Float16* sW2 = (_Float16*)(smem + 128 * 64 * 2);                 // [128][128]
  _Float16* sBuf = (_Float16*)(smem + 128 * 64 * 2 + 128 * 128 * 2);// per wave: attr + hid
  const int tid = threadIdx.x;
  for (int idx = tid; idx < 128 * 64; idx += blockDim.x) {
    int n = idx >> 6, k = idx & 63;
    sW1[idx] = (k < 50) ? (_Float16)w1[n * 50 + k] : (_Float16)0.0f;
  }
  for (int idx = tid; idx < 128 * 128; idx += blockDim.x)
    sW2[idx] = (_Float16)w2[idx];
  __syncthreads();
  const int w = tid >> 5, l = tid & 31;
  _Float16* sAttr = sBuf + w * (16 * 64 + 16 * 128);
  _Float16* sHid  = sAttr + 16 * 64;
  const int r = l & 15, sel = (l >> 4) & 1;
  const int mOff = (l & 16) ? 8 : 0;
  const int nl = l & 15;
  // hoist biases out of the tile loop (depend only on lane)
  float bias1[8], bias2[8];
#pragma unroll
  for (int nt = 0; nt < 8; ++nt) {
    bias1[nt] = b1[nt * 16 + nl];
    bias2[nt] = b2[nt * 16 + nl];
  }
  const int tiles = (E + 15) >> 4;
  for (int t = blockIdx.x * 8 + w; t < tiles; t += gridDim.x * 8) {
    const int e0 = t * 16;
    if (e0 + 256 < E)
      __builtin_prefetch(ea + (size_t)(e0 + 256) * 50 + l * 8, 0, 1);
    // stage edge_attr tile -> f16 LDS [16][64], zero-padded K=50..63
    // split 50 = 26 + 24 so every packed u32 store stays 4B-aligned
    {
      int e = e0 + r; if (e >= E) e = E - 1;
      const float2* arow = (const float2*)(ea + (size_t)e * 50 + sel * 26);
      if (sel == 0) {
#pragma unroll
        for (int i = 0; i < 13; ++i) {
          float2 v = arow[i];
          *(uint32_t*)(sAttr + r * 64 + i * 2) = pack2(v.x, v.y);
        }
      } else {
#pragma unroll
        for (int i = 0; i < 12; ++i) {
          float2 v = arow[i];
          *(uint32_t*)(sAttr + r * 64 + 26 + i * 2) = pack2(v.x, v.y);
        }
#pragma unroll
        for (int i = 0; i < 7; ++i)
          *(uint32_t*)(sAttr + r * 64 + 50 + i * 2) = 0u;
      }
    }
    // per-lane metadata for the 8 D-fragment rows this lane owns
    float cv[8]; int sR[8], dR[8];
#pragma unroll
    for (int j = 0; j < 8; ++j) {
      int e = e0 + mOff + j;
      if (e < E) {
        float wv = ew[e];
        cv[j] = 10.0f / (1e-10f + wv * wv) - 1.0f;
        sR[j] = ei[e];
        dR[j] = ei[E + e];
      } else { cv[j] = 0.0f; sR[j] = 0; dR[j] = 0; }
    }
    asm volatile("" ::: "memory");
    v16h a0 = frag_a_lds(sAttr, 64, 0);
    v16h a1 = frag_a_lds(sAttr, 64, 32);
    // hidden = ssp(attr @ W1^T + b1) -> f16 LDS [16][128]
#pragma unroll
    for (int nt = 0; nt < 8; ++nt) {
      v8f acc;
#pragma unroll
      for (int j = 0; j < 8; ++j) acc[j] = bias1[nt];
      acc = wmma_f16(a0, frag_b_lds(sW1, 64, nt * 16, 0), acc);
      acc = wmma_f16(a1, frag_b_lds(sW1, 64, nt * 16, 32), acc);
      int n = nt * 16 + nl;
#pragma unroll
      for (int j = 0; j < 8; ++j)
        sHid[(mOff + j) * 128 + n] = (_Float16)ssp_f(acc[j]);
    }
    asm volatile("" ::: "memory");
    v16h hK[4];
#pragma unroll
    for (int ks = 0; ks < 4; ++ks) hK[ks] = frag_a_lds(sHid, 128, ks * 32);
#pragma unroll
    for (int nt = 0; nt < 8; ++nt) {
      v8f acc;
#pragma unroll
      for (int j = 0; j < 8; ++j) acc[j] = bias2[nt];
#pragma unroll
      for (int ks = 0; ks < 4; ++ks)
        acc = wmma_f16(hK[ks], frag_b_lds(sW2, 128, nt * 16, ks * 32), acc);
      int n = nt * 16 + nl;
#pragma unroll
      for (int j = 0; j < 8; ++j) {
        float wv = acc[j] * cv[j];
        float hv = (float)hbuf[(size_t)n * N + sR[j]]; // feature-major gather (u16)
        atomicAdd(&agg[(size_t)dR[j] * 128 + n], hv * wv);
      }
    }
  }
}

// ---------------------------------------------------------------- kernel 4a
__global__ void cnt_kernel(const int* __restrict__ seg, float* __restrict__ cnt, int N) {
  int i = blockIdx.x * blockDim.x + threadIdx.x;
  int stride = gridDim.x * blockDim.x;
  for (; i < N; i += stride) atomicAdd(&cnt[seg[i]], 1.0f);
}

// ---------------------------------------------------------------- kernel 4b
__global__ void comp_reduce_kernel(const float* __restrict__ agg, const int* __restrict__ member,
                                   const int* __restrict__ seg, float* __restrict__ sums,
                                   int N) {
  long total = (long)N * 128;
  long i = blockIdx.x * (long)blockDim.x + threadIdx.x;
  long stride = (long)gridDim.x * blockDim.x;
  for (; i < total; i += stride) {
    int t = (int)(i >> 7), f = (int)(i & 127);
    int s = seg[t];
    atomicAdd(&sums[(size_t)s * 128 + f], agg[(size_t)member[t] * 128 + f]);
  }
}

// ---------------------------------------------------------------- kernel 5
// out = ssp(mean@lin2^T + b2) @ lin^T + b
__global__ void out_kernel(const float* __restrict__ sums, const float* __restrict__ cnt,
                           const float* __restrict__ lin2_w, const float* __restrict__ lin2_b,
                           const float* __restrict__ lin_w, const float* __restrict__ lin_b,
                           float* __restrict__ out, int C) {
  extern __shared__ char smem[];
  _Float16* sL2 = (_Float16*)smem;
  _Float16* sLW = (_Float16*)(smem + 128 * 128 * 2);
  _Float16* sBuf = (_Float16*)(smem + 2 * 128 * 128 * 2);
  const int tid = threadIdx.x;
  for (int idx = tid; idx < 128 * 128; idx += blockDim.x) {
    sL2[idx] = (_Float16)lin2_w[idx];
    sLW[idx] = (_Float16)lin_w[idx];
  }
  __syncthreads();
  const int w = tid >> 5, l = tid & 31;
  const int tiles = (C + 15) >> 4;
  int t = blockIdx.x * (blockDim.x >> 5) + w;
  if (t >= tiles) return;
  _Float16* sMean = sBuf + w * (2 * 16 * 128);
  _Float16* sAct  = sMean + 16 * 128;
  const int r = l & 15, sel = (l >> 4) & 1;
  const int mOff = (l & 16) ? 8 : 0;
  const int nl = l & 15;
  int c = t * 16 + r; if (c >= C) c = C - 1;
  float inv = 1.0f / fmaxf(cnt[c], 1.0f);
#pragma unroll
  for (int i = 0; i < 32; ++i) {
    int k = sel * 64 + i * 2;
    float2 v = *(const float2*)(sums + (size_t)c * 128 + k);
    *(uint32_t*)(sMean + r * 128 + k) = pack2(v.x * inv, v.y * inv);
  }
  asm volatile("" ::: "memory");
  v16h aK[4];
#pragma unroll
  for (int ks = 0; ks < 4; ++ks) aK[ks] = frag_a_lds(sMean, 128, ks * 32);
#pragma unroll
  for (int nt = 0; nt < 8; ++nt) {
    v8f acc;
#pragma unroll
    for (int j = 0; j < 8; ++j) acc[j] = lin2_b[nt * 16 + nl];
#pragma unroll
    for (int ks = 0; ks < 4; ++ks)
      acc = wmma_f16(aK[ks], frag_b_lds(sL2, 128, nt * 16, ks * 32), acc);
    int n = nt * 16 + nl;
#pragma unroll
    for (int j = 0; j < 8; ++j)
      sAct[(mOff + j) * 128 + n] = (_Float16)ssp_f(acc[j]);
  }
  asm volatile("" ::: "memory");
  v16h bK[4];
#pragma unroll
  for (int ks = 0; ks < 4; ++ks) bK[ks] = frag_a_lds(sAct, 128, ks * 32);
  const bool full = (t * 16 + 16 <= C); // uniform fast path
#pragma unroll
  for (int nt = 0; nt < 8; ++nt) {
    v8f acc;
#pragma unroll
    for (int j = 0; j < 8; ++j) acc[j] = lin_b[nt * 16 + nl];
#pragma unroll
    for (int ks = 0; ks < 4; ++ks)
      acc = wmma_f16(bK[ks], frag_b_lds(sLW, 128, nt * 16, ks * 32), acc);
    int n = nt * 16 + nl;
    if (full) {
#pragma unroll
      for (int j = 0; j < 8; ++j)
        out[(size_t)(t * 16 + mOff + j) * 128 + n] = acc[j];
    } else {
#pragma unroll
      for (int j = 0; j < 8; ++j) {
        int cc = t * 16 + mOff + j;
        if (cc < C) out[(size_t)cc * 128 + n] = acc[j];
      }
    }
  }
}

// ---------------------------------------------------------------- launcher
extern "C" void kernel_launch(void* const* d_in, const int* in_sizes, int n_in,
                              void* d_out, int out_size, void* d_ws, size_t ws_size,
                              hipStream_t stream) {
  const float* x   = (const float*)d_in[0];
  const int*   ei  = (const int*)d_in[1];
  const float* ew  = (const float*)d_in[2];
  const float* ea  = (const float*)d_in[3];
  const int*   mem = (const int*)d_in[4];
  const int*   seg = (const int*)d_in[5];
  const float* w1  = (const float*)d_in[6];
  const float* b1  = (const float*)d_in[7];
  const float* w2  = (const float*)d_in[8];
  const float* b2  = (const float*)d_in[9];
  const float* l1w = (const float*)d_in[10];
  const float* l2w = (const float*)d_in[11];
  const float* l2b = (const float*)d_in[12];
  const float* lw  = (const float*)d_in[13];
  const float* lb  = (const float*)d_in[14];
  (void)n_in; (void)ws_size;

  const int N = in_sizes[0] / 128;
  const int E = in_sizes[2];
  const int C = out_size / 128; // NUM_COMP = 2000

  char* ws = (char*)d_ws;
  _Float16* hbuf = (_Float16*)ws;                    // 128 x N f16 (feature-major)
  size_t off = ((size_t)N * 128 * 2 + 255) & ~(size_t)255;
  float* agg  = (float*)(ws + off);                  // N*128 f32
  float* sums = agg + (size_t)N * 128;               // C*128 f32
  float* cnt  = sums + (size_t)C * 128;              // C f32

  long zn = (long)N * 128 + (long)C * 128 + C;
  zero_kernel<<<2048, 256, 0, stream>>>(agg, zn);

  cnt_kernel<<<256, 256, 0, stream>>>(seg, cnt, N);

  size_t lds_h = 128 * 128 * 2 + 8 * 16 * 128 * 2;   // 64 KB
  int tilesN = (N + 15) >> 4;
  node_proj_kernel<<<(tilesN + 7) / 8, 256, lds_h, stream>>>(x, l1w, hbuf, N);

  size_t lds_e = 128 * 64 * 2 + 128 * 128 * 2 + 8 * (16 * 64 + 16 * 128) * 2; // 96 KB
  edge_kernel<<<2500, 256, lds_e, stream>>>(ea, ei, ew, w1, b1, w2, b2, hbuf, agg, E, N);

  comp_reduce_kernel<<<2048, 256, 0, stream>>>(agg, mem, seg, sums, N);

  size_t lds_o = 2 * 128 * 128 * 2 + 4 * 2 * 16 * 128 * 2; // 96 KB
  int tilesC = (C + 15) >> 4;
  out_kernel<<<(tilesC + 3) / 4, 128, lds_o, stream>>>(sums, cnt, l2w, l2b, lw, lb,
                                                       (float*)d_out, C);
}
